// Attention_layer_32349693673756
// MI455X (gfx1250) — matrix operand order repaired and checked by reading, not compiled, not run on hardware
//
#include <hip/hip_runtime.h>

#define F_IN  128
#define F_OUT 32
#define NHEAD 4
#define DEG   16

typedef __attribute__((ext_vector_type(16))) _Float16 v16h;
typedef __attribute__((ext_vector_type(8)))  float    v8f;

// ---------------------------------------------------------------------------
// Kernel 1: t[h][n][o] = sum_k x[n][k] * Ws[h][o][k]
// f16 WMMA (16x16x32), f32 accumulate. One wave = 16-node x 32-feature tile
// for one head. Block = 256 threads = 8 waves = 32 nodes x 4 heads.
// ---------------------------------------------------------------------------
__global__ __launch_bounds__(256) void gat_gemm_wmma(
    const float* __restrict__ x, const float* __restrict__ Ws,
    float* __restrict__ t, int n)
{
    // W transposed to [head][k][o] in f16: B-fragment rows are contiguous 64B.
    __shared__ __align__(64) _Float16 Wt[NHEAD][F_IN][F_OUT];   // 32 KB

    const int tid = threadIdx.x;
    for (int e = tid; e < NHEAD * F_OUT * F_IN; e += 256) {
        const int h   = e >> 12;        // 4096 elems per head
        const int rem = e & 4095;
        const int o   = rem >> 7;       // 0..31
        const int k   = rem & 127;      // 0..127
        Wt[h][k][o] = (_Float16)Ws[e];
    }
    __syncthreads();

    const int wave      = tid >> 5;
    const int lane      = tid & 31;
    const int head      = wave & 3;
    const int nodeBase  = blockIdx.x * 32 + (wave >> 2) * 16;
    const int laneHalf  = lane >> 4;      // A-frag: K offset 0 or 8
    const int rowInTile = lane & 15;      // A-frag: M = lane % 16

    int row = nodeBase + rowInTile;
    if (row >= n) row = n - 1;            // clamp (stores are guarded)
    const float* __restrict__ xrow = x + (size_t)row * F_IN;

    v8f acc0 = {};   // output cols 0..15
    v8f acc1 = {};   // output cols 16..31

#pragma unroll
    for (int kt = 0; kt < 4; ++kt) {
        // A fragment (16x32 f16): per ISA layout, lane holds row M=lane%16,
        // K pairs {c..c+7} in v0..v3 and {c+16..c+23} in v4..v7, c = 8*laneHalf.
        const int c = kt * 32 + laneHalf * 8;
        const float4 f0 = *(const float4*)(xrow + c);
        const float4 f1 = *(const float4*)(xrow + c + 4);
        const float4 f2 = *(const float4*)(xrow + c + 16);
        const float4 f3 = *(const float4*)(xrow + c + 20);
        v16h a;
        a[0]=(_Float16)f0.x;  a[1]=(_Float16)f0.y;  a[2]=(_Float16)f0.z;  a[3]=(_Float16)f0.w;
        a[4]=(_Float16)f1.x;  a[5]=(_Float16)f1.y;  a[6]=(_Float16)f1.z;  a[7]=(_Float16)f1.w;
        a[8]=(_Float16)f2.x;  a[9]=(_Float16)f2.y;  a[10]=(_Float16)f2.z; a[11]=(_Float16)f2.w;
        a[12]=(_Float16)f3.x; a[13]=(_Float16)f3.y; a[14]=(_Float16)f3.z; a[15]=(_Float16)f3.w;

        // B fragments (32x16 f16): lane holds column N = lane%16, K in the same
        // order as A: {c..c+7} then {c+16..c+23}, c = kt*32 + 8*laneHalf.
        const int bc = lane & 15;
        v16h b0, b1;
#pragma unroll
        for (int s = 0; s < 16; ++s) {
            const int kk = kt * 32 + laneHalf * 8 + (s < 8 ? s : 8 + s);
            b0[s] = Wt[head][kk][bc];
            b1[s] = Wt[head][kk][16 + bc];
        }

        acc0 = __builtin_amdgcn_wmma_f32_16x16x32_f16(
                   false, a, false, b0, (short)0, acc0, false, false);
        acc1 = __builtin_amdgcn_wmma_f32_16x16x32_f16(
                   false, a, false, b1, (short)0, acc1, false, false);
    }

    // C/D layout: VGPR v holds row M = v + 8*laneHalf, col N = lane%16.
    const int col = lane & 15;
#pragma unroll
    for (int v = 0; v < 8; ++v) {
        const int r = nodeBase + v + 8 * laneHalf;
        if (r < n) {
            const size_t base = ((size_t)head * n + r) * F_OUT;
            t[base + col]      = acc0[v];
            t[base + 16 + col] = acc1[v];
        }
    }
}

// ---------------------------------------------------------------------------
// Kernel 2: s_src[h][n] = t[h][n][:].a_src[h],  s_dst likewise.
// One thread per (h,n); 128B contiguous reads of L2-resident t.
// ---------------------------------------------------------------------------
__global__ __launch_bounds__(256) void gat_scores(
    const float* __restrict__ t, const float* __restrict__ As,
    float* __restrict__ s_src, float* __restrict__ s_dst, int n)
{
    const int gid = blockIdx.x * blockDim.x + threadIdx.x;   // h*n + node
    if (gid >= NHEAD * n) return;
    const int h = gid / n;

    const float* __restrict__ trow = t + (size_t)gid * F_OUT;
    const float* __restrict__ a    = As + h * 2 * F_OUT;     // [src(32) | dst(32)]

    float ss = 0.f, sd = 0.f;
#pragma unroll
    for (int f = 0; f < F_OUT; ++f) {
        const float v = trow[f];
        ss += v * a[f];
        sd += v * a[F_OUT + f];
    }
    s_src[gid] = ss;
    s_dst[gid] = sd;
}

// ---------------------------------------------------------------------------
// Kernel 3: per (node, head) wave. Lanes 0..16 build scores, wave-wide
// softmax of e = exp(leaky_relu(score)) (matches reference exactly), then
// lane=f accumulates the alpha-weighted gather over 17 rows of t, ELU, store.
// ---------------------------------------------------------------------------
__global__ __launch_bounds__(256) void gat_aggregate(
    const int* __restrict__ neigh, const float* __restrict__ t,
    const float* __restrict__ s_src, const float* __restrict__ s_dst,
    float* __restrict__ out, int n)
{
    const int gwave = (blockIdx.x * blockDim.x + threadIdx.x) >> 5;
    const int lane  = threadIdx.x & 31;
    const int node  = gwave >> 2;
    const int head  = gwave & 3;
    if (node >= n) return;

    // lane d in [0,16]: d==0 is self, else neighbor d-1
    int idx = node;
    if (lane >= 1 && lane <= DEG) idx = neigh[node * DEG + (lane - 1)];

    float e;
    if (lane <= DEG) {
        const float sc = s_src[(size_t)head * n + node] +
                         s_dst[(size_t)head * n + idx];
        const float lr = sc > 0.f ? sc : 0.2f * sc;
        e = __expf(lr);
    } else {
        e = -__builtin_inff();
    }

    // softmax over the 17 valid lanes (stable, like jax.nn.softmax)
    float m = e;
#pragma unroll
    for (int off = 16; off > 0; off >>= 1)
        m = fmaxf(m, __shfl_xor(m, off, 32));
    float p = (lane <= DEG) ? __expf(e - m) : 0.f;
    float s = p;
#pragma unroll
    for (int off = 16; off > 0; off >>= 1)
        s += __shfl_xor(s, off, 32);
    const float alpha = p / s;

    // lane = output feature f; broadcast alpha_d / idx_d, gather 128B rows
    float acc = 0.f;
#pragma unroll
    for (int d = 0; d <= DEG; ++d) {
        const float ad = __shfl(alpha, d, 32);
        const int   id = __shfl(idx,   d, 32);
        acc += ad * t[((size_t)head * n + id) * F_OUT + lane];
    }

    const float o = acc > 0.f ? acc : (__expf(acc) - 1.f);   // ELU
    out[(size_t)node * (NHEAD * F_OUT) + head * F_OUT + lane] = o;
}

// ---------------------------------------------------------------------------
extern "C" void kernel_launch(void* const* d_in, const int* in_sizes, int n_in,
                              void* d_out, int out_size, void* d_ws, size_t ws_size,
                              hipStream_t stream)
{
    const float* x     = (const float*)d_in[0];   // (N, 128) f32
    const int*   neigh = (const int*)  d_in[1];   // (N, 16)  i32
    const float* Ws    = (const float*)d_in[2];   // (4, 32, 128) f32
    const float* As    = (const float*)d_in[3];   // (4, 64, 1)   f32
    float*       out   = (float*)d_out;           // (N, 128) f32

    const int n = in_sizes[0] / F_IN;

    float* t     = (float*)d_ws;                          // H*N*32 floats
    float* s_src = t + (size_t)NHEAD * n * F_OUT;         // H*N floats
    float* s_dst = s_src + (size_t)NHEAD * n;             // H*N floats

    // K1: 8 waves/block, 32 nodes x 4 heads per block
    const int gemmBlocks = (n + 31) / 32;
    gat_gemm_wmma<<<gemmBlocks, 256, 0, stream>>>(x, Ws, t, n);

    // K2: one thread per (h,n)
    const int scoreBlocks = (NHEAD * n + 255) / 256;
    gat_scores<<<scoreBlocks, 256, 0, stream>>>(t, As, s_src, s_dst, n);

    // K3: one wave per (node, head) -> n*4 waves, 8 waves/block
    const int aggBlocks = (n * NHEAD + 7) / 8;
    gat_aggregate<<<aggBlocks, 256, 0, stream>>>(neigh, t, s_src, s_dst, out, n);
}
